// MultiHeadAttention_13280038879368
// MI455X (gfx1250) — matrix operand order repaired and checked by reading, compiled, not run on hardware
//
#include <hip/hip_runtime.h>

// ---------------- CDNA5 (gfx1250) fused MHA, bf16 WMMA pipeline ----------------
// wave32; v_wmma_f32_16x16x32_bf16 for all GEMMs (QKV proj, QK^T, PV, out proj).

typedef __attribute__((ext_vector_type(16))) __bf16 bf16x16;
typedef __attribute__((ext_vector_type(8)))  float  f32x8;

constexpr int Bv = 2, Sv = 2048, Dv = 1024, Hv = 16, HDv = 64;
constexpr int NROW = Bv * Sv;              // 4096 total rows (b,s)
constexpr float SM_SCALE = 0.125f;         // 1/sqrt(64)
constexpr float NEG_BIG  = -3.0e38f;

#define DEV static __device__ __forceinline__

DEV bf16x16 ldfrag(const __bf16* p) { return *(const bf16x16*)p; }

DEV f32x8 wmma_bf16(bf16x16 a, bf16x16 b, f32x8 c) {
  return __builtin_amdgcn_wmma_f32_16x16x32_bf16(
      /*neg_a=*/false, a, /*neg_b=*/false, b,
      /*c_mod=*/(short)0, c, /*reuse_a=*/false, /*reuse_b=*/false);
}

// A-fragment (16x32 bf16) element k for (lane, i); lane holds row m=lane&15.
DEV int a_k_of(int lane, int i) {
  return (i < 8) ? ((lane < 16 ? 0 : 8) + i)
                 : ((lane < 16 ? 16 : 24) + (i - 8));
}
// B-fragment (32x16 bf16) element k for (lane, i); lane holds col n=lane&15.
DEV int b_k_of(int lane, int i) {
  return (i < 8) ? ((lane < 16 ? 0 : 8) + i)
                 : ((lane < 16 ? 16 : 24) + (i - 8));
}

// ---------------- pack kernels ----------------
// A-pack: src [rows x cols] row-major (f32) -> fragment tiles, 32B/lane.
__global__ void k_pack_a_f32(const float* __restrict__ src, int rows, int cols,
                             int ld, __bf16* __restrict__ dst) {
  int idx = blockIdx.x * blockDim.x + threadIdx.x;
  int total = (rows >> 4) * (cols >> 5) * 32;
  if (idx >= total) return;
  int lane = idx & 31, tile = idx >> 5;
  int tkn = cols >> 5;
  int ti = tile / tkn, tk = tile % tkn;
  int m = lane & 15;
  const float* s = src + (size_t)(ti * 16 + m) * ld + tk * 32;
  __bf16* d = dst + (size_t)idx * 16;
#pragma unroll
  for (int i = 0; i < 16; ++i) d[i] = (__bf16)s[a_k_of(lane, i)];
}

__global__ void k_pack_a_bf16(const __bf16* __restrict__ src, int rows, int cols,
                              int ld, __bf16* __restrict__ dst) {
  int idx = blockIdx.x * blockDim.x + threadIdx.x;
  int total = (rows >> 4) * (cols >> 5) * 32;
  if (idx >= total) return;
  int lane = idx & 31, tile = idx >> 5;
  int tkn = cols >> 5;
  int ti = tile / tkn, tk = tile % tkn;
  int m = lane & 15;
  const __bf16* s = src + (size_t)(ti * 16 + m) * ld + tk * 32;
  __bf16* d = dst + (size_t)idx * 16;
#pragma unroll
  for (int i = 0; i < 16; ++i) d[i] = s[a_k_of(lane, i)];
}

// B-pack of W^T: B[k,n] = W[o0+n, k0+k]  (torch Linear: out = x @ W.T)
__global__ void k_pack_bT_f32(const float* __restrict__ W, int O, int Kd,
                              int ld, __bf16* __restrict__ dst) {
  int idx = blockIdx.x * blockDim.x + threadIdx.x;
  int total = (O >> 4) * (Kd >> 5) * 32;
  if (idx >= total) return;
  int lane = idx & 31, tile = idx >> 5;
  int tkn = Kd >> 5;
  int to = tile / tkn, tk = tile % tkn;
  int n = lane & 15;
  const float* s = W + (size_t)(to * 16 + n) * ld + tk * 32;
  __bf16* d = dst + (size_t)idx * 16;
#pragma unroll
  for (int i = 0; i < 16; ++i) d[i] = (__bf16)s[b_k_of(lane, i)];
}

// per-(b,h) packs for attention; grid.y = b*H + h
__global__ void k_pack_q(const __bf16* __restrict__ Qrm, __bf16* __restrict__ Qp) {
  int bh = blockIdx.y; int b = bh / Hv, h = bh % Hv;
  int idx = blockIdx.x * blockDim.x + threadIdx.x;   // 8192 per bh
  int lane = idx & 31, tile = idx >> 5;              // tile = ti*2 + tk
  int ti = tile >> 1, tk = tile & 1;
  int m = lane & 15;
  const __bf16* s = Qrm + (size_t)(b * Sv + ti * 16 + m) * Dv + h * HDv + tk * 32;
  __bf16* d = Qp + (size_t)bh * (Sv * HDv) + (size_t)idx * 16;
#pragma unroll
  for (int i = 0; i < 16; ++i) d[i] = s[a_k_of(lane, i)];
}

// K as B^T-pack per head: B[k=d, n=key] = K[key, d]; tile = keytile*2 + dtile32
__global__ void k_pack_k(const __bf16* __restrict__ Krm, __bf16* __restrict__ Kp) {
  int bh = blockIdx.y; int b = bh / Hv, h = bh % Hv;
  int idx = blockIdx.x * blockDim.x + threadIdx.x;
  int lane = idx & 31, tile = idx >> 5;
  int to = tile >> 1, tk = tile & 1;                 // to: key 16-tile, tk: d 32-tile
  int n = lane & 15;
  const __bf16* s = Krm + (size_t)(b * Sv + to * 16 + n) * Dv + h * HDv + tk * 32;
  __bf16* d = Kp + (size_t)bh * (Sv * HDv) + (size_t)idx * 16;
#pragma unroll
  for (int i = 0; i < 16; ++i) d[i] = s[b_k_of(lane, i)];
}

// V as direct B-pack per head: B[k=key, n=d] = V[key, d]; tile = dtile16*64 + keytile32
__global__ void k_pack_v(const __bf16* __restrict__ Vrm, __bf16* __restrict__ Vp) {
  int bh = blockIdx.y; int b = bh / Hv, h = bh % Hv;
  int idx = blockIdx.x * blockDim.x + threadIdx.x;
  int lane = idx & 31, tile = idx >> 5;
  int to = tile >> 6, tk = tile & 63;                // to: d 16-tile (0..3), tk: key 32-tile
  int n = lane & 15;
  __bf16* d = Vp + (size_t)bh * (Sv * HDv) + (size_t)idx * 16;
#pragma unroll
  for (int i = 0; i < 16; ++i) {
    int k = b_k_of(lane, i);                         // key within 32-block
    d[i] = Vrm[(size_t)(b * Sv + tk * 32 + k) * Dv + h * HDv + to * 16 + n];
  }
}

// ---------------- WMMA GEMM: C[N,O] = A[N,K] * W^T, out bf16 or f32+bias ------
__global__ void __launch_bounds__(256)
k_gemm(const __bf16* __restrict__ Ap, const __bf16* __restrict__ Bp,
       int Nr, int Oc, int Kd,
       __bf16* __restrict__ outB, float* __restrict__ outF,
       const float* __restrict__ bias) {
  int lane = threadIdx.x;
  int w = blockIdx.x * blockDim.y + threadIdx.y;
  int oT2 = Oc >> 5;
  int tn = w / oT2, to = w % oT2;
  int Kt = Kd >> 5;
  f32x8 acc[2][2] = {};
  const __bf16* a0p = Ap + ((size_t)(tn * 2 + 0) * Kt * 32 + lane) * 16;
  const __bf16* a1p = Ap + ((size_t)(tn * 2 + 1) * Kt * 32 + lane) * 16;
  const __bf16* b0p = Bp + ((size_t)(to * 2 + 0) * Kt * 32 + lane) * 16;
  const __bf16* b1p = Bp + ((size_t)(to * 2 + 1) * Kt * 32 + lane) * 16;
  for (int kt = 0; kt < Kt; ++kt) {
    bf16x16 a0 = ldfrag(a0p + (size_t)kt * 512);
    bf16x16 a1 = ldfrag(a1p + (size_t)kt * 512);
    bf16x16 b0 = ldfrag(b0p + (size_t)kt * 512);
    bf16x16 b1 = ldfrag(b1p + (size_t)kt * 512);
    acc[0][0] = wmma_bf16(a0, b0, acc[0][0]);
    acc[0][1] = wmma_bf16(a0, b1, acc[0][1]);
    acc[1][0] = wmma_bf16(a1, b0, acc[1][0]);
    acc[1][1] = wmma_bf16(a1, b1, acc[1][1]);
  }
  int nl = lane & 15, moff = (lane & 16) ? 8 : 0;
  if (outF) {
#pragma unroll
    for (int i = 0; i < 2; ++i)
#pragma unroll
      for (int j = 0; j < 2; ++j)
#pragma unroll
        for (int r = 0; r < 8; ++r) {
          size_t row = (size_t)(tn * 32 + i * 16 + r + moff);
          size_t col = (size_t)(to * 32 + j * 16 + nl);
          outF[row * Oc + col] = acc[i][j][r] + bias[col];
        }
  } else {
#pragma unroll
    for (int i = 0; i < 2; ++i)
#pragma unroll
      for (int j = 0; j < 2; ++j)
#pragma unroll
        for (int r = 0; r < 8; ++r) {
          size_t row = (size_t)(tn * 32 + i * 16 + r + moff);
          size_t col = (size_t)(to * 32 + j * 16 + nl);
          outB[row * Oc + col] = (__bf16)acc[i][j][r];
        }
  }
}

// ---------------- flash attention: 1 wave per (b,h, 16-row q tile) ------------
__global__ void __launch_bounds__(256)
k_attn(const __bf16* __restrict__ Qp, const __bf16* __restrict__ Kp,
       const __bf16* __restrict__ Vp, __bf16* __restrict__ ctx) {
  __shared__ __align__(32) __bf16 smem[8 * 512];     // 1KB per wave: P tile 16x32
  int lane = threadIdx.x, ty = threadIdx.y;
  int w = blockIdx.x * 8 + ty;
  int bh = w >> 7;                                   // 128 q-tiles per (b,h)
  int qt = w & 127;
  int b = bh >> 4, h = bh & 15;
  const __bf16* Qb = Qp + (size_t)bh * (Sv * HDv);
  const __bf16* Kb = Kp + (size_t)bh * (Sv * HDv);
  const __bf16* Vb = Vp + (size_t)bh * (Sv * HDv);

  bf16x16 qa0 = ldfrag(Qb + ((size_t)(qt * 2 + 0) * 32 + lane) * 16);
  bf16x16 qa1 = ldfrag(Qb + ((size_t)(qt * 2 + 1) * 32 + lane) * 16);

  f32x8 acc[4] = {};
  float accM[8], accL[8];
#pragma unroll
  for (int r = 0; r < 8; ++r) { accM[r] = NEG_BIG; accL[r] = 0.f; }

  int nl = lane & 15, moff = (lane & 16) ? 8 : 0;
  int nkt = qt + 1;                                  // causal: key 16-tiles
  int nkt2 = (nkt + 1) >> 1;                         // 32-key steps
  __bf16* sm = smem + ty * 512;

  for (int kt2 = 0; kt2 < nkt2; ++kt2) {
    float sv[2][8];
#pragma unroll
    for (int s = 0; s < 2; ++s) {
      int kti = kt2 * 2 + s;
      if (kti < nkt) {                               // wave-uniform branch
        f32x8 sc = {};
        sc = wmma_bf16(qa0, ldfrag(Kb + ((size_t)(kti * 2 + 0) * 32 + lane) * 16), sc);
        sc = wmma_bf16(qa1, ldfrag(Kb + ((size_t)(kti * 2 + 1) * 32 + lane) * 16), sc);
        int key = kti * 16 + nl;
#pragma unroll
        for (int r = 0; r < 8; ++r) {
          int qrow = qt * 16 + r + moff;
          sv[s][r] = (key <= qrow) ? sc[r] * SM_SCALE : NEG_BIG;
        }
      } else {
#pragma unroll
        for (int r = 0; r < 8; ++r) sv[s][r] = NEG_BIG;
      }
    }
    // online softmax: row max / row sum across 16 lanes of each half
    float tmax[8];
#pragma unroll
    for (int r = 0; r < 8; ++r) tmax[r] = fmaxf(sv[0][r], sv[1][r]);
#pragma unroll
    for (int d2 = 1; d2 < 16; d2 <<= 1)
#pragma unroll
      for (int r = 0; r < 8; ++r) tmax[r] = fmaxf(tmax[r], __shfl_xor(tmax[r], d2));
    float alpha[8];
#pragma unroll
    for (int r = 0; r < 8; ++r) {
      float nm = fmaxf(accM[r], tmax[r]);
      alpha[r] = __expf(accM[r] - nm);
      accM[r] = nm;
    }
    float rs[8];
#pragma unroll
    for (int r = 0; r < 8; ++r) {
      sv[0][r] = __expf(sv[0][r] - accM[r]);
      sv[1][r] = __expf(sv[1][r] - accM[r]);
      rs[r] = sv[0][r] + sv[1][r];
    }
#pragma unroll
    for (int d2 = 1; d2 < 16; d2 <<= 1)
#pragma unroll
      for (int r = 0; r < 8; ++r) rs[r] += __shfl_xor(rs[r], d2);
#pragma unroll
    for (int r = 0; r < 8; ++r) accL[r] = accL[r] * alpha[r] + rs[r];
#pragma unroll
    for (int dt = 0; dt < 4; ++dt)
#pragma unroll
      for (int r = 0; r < 8; ++r) acc[dt][r] *= alpha[r];

    // P (C-layout f32) -> LDS scatter in A-fragment order, then vector reload
#pragma unroll
    for (int s = 0; s < 2; ++s) {
      int kk = s * 16 + nl;
      int lp16 = (kk & 8) ? 16 : 0;
      int ii = ((kk & 16) ? 8 : 0) + (kk & 7);
#pragma unroll
      for (int r = 0; r < 8; ++r) {
        int m = r + moff;
        sm[(m + lp16) * 16 + ii] = (__bf16)sv[s][r];
      }
    }
    asm volatile("s_wait_dscnt 0" ::: "memory");     // wave-local LDS RAW fence
    bf16x16 pf = ldfrag(sm + lane * 16);
#pragma unroll
    for (int dt = 0; dt < 4; ++dt) {
      bf16x16 vb = ldfrag(Vb + ((size_t)(dt * 64 + kt2) * 32 + lane) * 16);
      acc[dt] = wmma_bf16(pf, vb, acc[dt]);
    }
  }

  // epilogue: normalize and write ctx (head-concat row-major bf16)
  float inv[8];
#pragma unroll
  for (int r = 0; r < 8; ++r) inv[r] = 1.0f / accL[r];
#pragma unroll
  for (int dt = 0; dt < 4; ++dt)
#pragma unroll
    for (int r = 0; r < 8; ++r) {
      size_t row = (size_t)(b * Sv + qt * 16 + r + moff);
      size_t col = (size_t)(h * HDv + dt * 16 + nl);
      ctx[row * Dv + col] = (__bf16)(acc[dt][r] * inv[r]);
    }
}

// ---------------- host orchestration ----------------
extern "C" void kernel_launch(void* const* d_in, const int* in_sizes, int n_in,
                              void* d_out, int out_size, void* d_ws, size_t ws_size,
                              hipStream_t stream) {
  (void)in_sizes; (void)n_in; (void)out_size; (void)ws_size;
  const float* x  = (const float*)d_in[0];
  const float* Wq = (const float*)d_in[1];
  const float* Wk = (const float*)d_in[2];
  const float* Wv = (const float*)d_in[3];
  const float* Wo = (const float*)d_in[4];
  const float* bo = (const float*)d_in[5];
  float* out = (float*)d_out;

  __bf16* ws = (__bf16*)d_ws;
  const size_t M1 = (size_t)1 << 20;                 // 1M bf16 elems
  __bf16* xA    = ws;                                //  4M (also reused as ctxA)
  __bf16* WqP   = ws + 4 * M1;                       //  1M each
  __bf16* WkP   = ws + 5 * M1;
  __bf16* WvP   = ws + 6 * M1;
  __bf16* WoP   = ws + 7 * M1;
  __bf16* Qrm   = ws + 8 * M1;                       //  4M (reused as ctx_rm)
  __bf16* Krm   = ws + 12 * M1;                      //  4M
  __bf16* Vrm   = ws + 16 * M1;                      //  4M
  __bf16* Qp    = ws + 20 * M1;                      //  4M
  __bf16* Kp    = ws + 24 * M1;                      //  4M
  __bf16* Vp    = ws + 28 * M1;                      //  4M  => 64 MiB total
  __bf16* ctxRm = Qrm;
  __bf16* ctxA  = xA;

  dim3 blk256(256);
  dim3 gblk(32, 8);                                  // 8 wave32 per block

  // 1) convert + pack inputs into WMMA fragment layouts
  k_pack_a_f32 <<<1024, blk256, 0, stream>>>(x,  NROW, Dv, Dv, xA);
  k_pack_bT_f32<<< 256, blk256, 0, stream>>>(Wq, Dv, Dv, Dv, WqP);
  k_pack_bT_f32<<< 256, blk256, 0, stream>>>(Wk, Dv, Dv, Dv, WkP);
  k_pack_bT_f32<<< 256, blk256, 0, stream>>>(Wv, Dv, Dv, Dv, WvP);
  k_pack_bT_f32<<< 256, blk256, 0, stream>>>(Wo, Dv, Dv, Dv, WoP);

  // 2) QKV projections (WMMA GEMMs, bf16 out)
  k_gemm<<<512, gblk, 0, stream>>>(xA, WqP, NROW, Dv, Dv, Qrm, nullptr, nullptr);
  k_gemm<<<512, gblk, 0, stream>>>(xA, WkP, NROW, Dv, Dv, Krm, nullptr, nullptr);
  k_gemm<<<512, gblk, 0, stream>>>(xA, WvP, NROW, Dv, Dv, Vrm, nullptr, nullptr);

  // 3) per-head fragment packs for attention
  dim3 pgrid(32, Bv * Hv);
  k_pack_q<<<pgrid, blk256, 0, stream>>>(Qrm, Qp);
  k_pack_k<<<pgrid, blk256, 0, stream>>>(Krm, Kp);
  k_pack_v<<<pgrid, blk256, 0, stream>>>(Vrm, Vp);

  // 4) causal flash attention (WMMA QK^T + WMMA PV, online softmax)
  k_attn<<<512, gblk, 0, stream>>>(Qp, Kp, Vp, ctxRm);

  // 5) output projection with bias (f32 out)
  k_pack_a_bf16<<<1024, blk256, 0, stream>>>(ctxRm, NROW, Dv, Dv, ctxA);
  k_gemm<<<512, gblk, 0, stream>>>(ctxA, WoP, NROW, Dv, Dv, nullptr, out, bo);
}
